// Conv2d_NN_6536940224927
// MI455X (gfx1250) — matrix operand order repaired and checked by reading, not compiled, not run on hardware
//
#include <hip/hip_runtime.h>
#include <math.h>

// Problem constants (from reference):
//   x: (32, 128, 32, 32) f32 ; W: (128, 128, 9) f32 ; out: (32, 128, 32, 32) f32
#define B_   32
#define C_   128
#define N_   1024          // 32*32 tokens
#define K_   9
#define O_   128
#define KPAD 16
#define INF_  1.0e10f
#define NEGF  (-3.0e38f)

typedef __attribute__((ext_vector_type(16))) _Float16     v16h;
typedef __attribute__((ext_vector_type(8)))  float        v8f;
typedef __attribute__((ext_vector_type(4)))  unsigned int v4u;

union Frag { v16h h; v4u u[2]; };

// CDNA5 async copy: per-lane 16B global -> LDS, tracked by ASYNCcnt.
__device__ __forceinline__ void async_g2l_b128(unsigned int lds_off,
                                               const _Float16* g) {
  asm volatile("global_load_async_to_lds_b128 %0, %1, off"
               :: "v"(lds_off), "v"((unsigned long long)(uintptr_t)g)
               : "memory");
}
__device__ __forceinline__ void wait_asynccnt0() {
  asm volatile("s_wait_asynccnt 0x0" ::: "memory");
}

// ---------------------------------------------------------------------------
// Kernel 1: per-token L2 normalize + f16 cast, token-major layout (B*N, C).
// 64 tokens per block; coalesced f32 reads, LDS transpose, b128 f16 writes.
// ---------------------------------------------------------------------------
__global__ __launch_bounds__(256) void k_normalize(
    const float* __restrict__ x, _Float16* __restrict__ xn,
    _Float16* __restrict__ xfh) {
  __shared__ __align__(16) _Float16 sxn[64][C_];
  __shared__ __align__(16) _Float16 sxf[64][C_];
  __shared__ float sss[4][64];
  __shared__ float srn[64];

  const int tid  = threadIdx.x;
  const int tok  = tid & 63;        // token within block
  const int grp  = tid >> 6;        // channel group: grp*32 .. +31
  const int b    = blockIdx.x >> 4; // 16 blocks per batch
  const int base = (blockIdx.x & 15) << 6;
  const int n    = base + tok;

  const float* xb = x + ((size_t)b * C_) * N_;
  float va[32];
  float ss = 0.f;
#pragma unroll
  for (int j = 0; j < 32; ++j) {
    const float v = xb[(size_t)(grp * 32 + j) * N_ + n];
    va[j] = v;
    ss += v * v;
  }
  sss[grp][tok] = ss;
  __syncthreads();
  if (tid < 64) {
    const float s = sss[0][tid] + sss[1][tid] + sss[2][tid] + sss[3][tid];
    srn[tid] = 1.0f / (sqrtf(s) + 1e-8f);
  }
  __syncthreads();
  const float rn = srn[tok];
#pragma unroll
  for (int j = 0; j < 32; ++j) {
    sxf[tok][grp * 32 + j] = (_Float16)va[j];
    sxn[tok][grp * 32 + j] = (_Float16)(va[j] * rn);
  }
  __syncthreads();
  // Coalesced b128 writes of the block's contiguous 16 KB region (each array).
  const v4u* pn = (const v4u*)&sxn[0][0];
  const v4u* pf = (const v4u*)&sxf[0][0];
  v4u* gn = (v4u*)(xn  + ((size_t)(b * N_ + base)) * C_);
  v4u* gf = (v4u*)(xfh + ((size_t)(b * N_ + base)) * C_);
#pragma unroll
  for (int j = 0; j < 4; ++j) {
    gn[tid + j * 256] = pn[tid + j * 256];
    gf[tid + j * 256] = pf[tid + j * 256];
  }
}

// ---------------------------------------------------------------------------
// Kernel 1b: swizzle W (O,C,K) f32 into WMMA-B-fragment-ordered f16:
//   wsw[kk][otile][lane][16 halves], kk in [0,36) over CK=1152, q = k*128+c.
// ---------------------------------------------------------------------------
__global__ __launch_bounds__(256) void k_wswizzle(
    const float* __restrict__ Wg, _Float16* __restrict__ wsw) {
  const int e    = blockIdx.x * 256 + threadIdx.x;  // < 36*8*32*16 = 147456
  const int h    = e & 15;
  const int lane = (e >> 4) & 31;
  const int ot   = (e >> 9) & 7;
  const int kk   = e >> 12;
  const int o    = ot * 16 + (lane & 15);
  const int q    = kk * 32 + (h & 8) * 2 + (lane >> 4) * 8 + (h & 7);
  const int k    = q >> 7;
  const int c    = q & 127;
  wsw[e] = (_Float16)Wg[(size_t)(o * C_ + c) * K_ + k];
}

// ---------------------------------------------------------------------------
// Kernel 2: fused cosine-sim GEMM (WMMA f16) + diag INF + top-9 selection.
// Block = 512 threads (16 waves); owns batch b, 16-row strip m0..m0+15.
// B fragments are double-buffered ACROSS chunks: chunk c+1's 8 b128 loads
// issue before chunk c's WMMA chain and their latency is hidden under the
// top-K VALU phase of chunk c. Chunk loop unrolled so buffers SSA-rename.
// ---------------------------------------------------------------------------
__device__ __forceinline__ void loadB4(Frag bf[4], const _Float16* base) {
#pragma unroll
  for (int j = 0; j < 4; ++j) {
    bf[j].u[0] = *(const v4u*)(base + j * 32);
    bf[j].u[1] = *(const v4u*)(base + j * 32 + 8);
  }
}

__global__ __launch_bounds__(512) void k_sim_topk(
    const _Float16* __restrict__ xn, int* __restrict__ idxws) {
  __shared__ float strip[16][265];   // 265 stride avoids bank conflicts
  __shared__ float rvs[16][KPAD];
  __shared__ int   ris[16][KPAD];
  __shared__ float cwv[16][9];
  __shared__ int   cwi[16][9];

  const int tid  = threadIdx.x;
  const int wave = tid >> 5;
  const int lane = tid & 31;
  const int b    = blockIdx.x >> 6;        // 64 strips per batch
  const int m0   = (blockIdx.x & 63) << 4;

  if (tid < 16 * KPAD) {
    rvs[tid >> 4][tid & 15] = NEGF;
    ris[tid >> 4][tid & 15] = 0x7fffffff;
  }
  __syncthreads();

  const int hp  = lane >> 4;   // half-wave select
  const int l16 = lane & 15;

  // A fragments: 16x32 f16 per K-chunk, contiguous k = 16*hp + s (same order as B).
  const _Float16* arow = xn + ((size_t)(b * N_ + m0 + l16)) * C_;
  Frag a[4];
#pragma unroll
  for (int j = 0; j < 4; ++j) {
    const int c0 = j * 32;
    a[j].u[0] = *(const v4u*)(arow + c0 + hp * 16);
    a[j].u[1] = *(const v4u*)(arow + c0 + hp * 16 + 8);
  }

  const _Float16* xnb = xn + ((size_t)b * N_) * C_;
  Frag bf[4];
  loadB4(bf, xnb + (size_t)((wave << 4) + l16) * C_ + hp * 16);  // chunk 0

#pragma unroll
  for (int chunk = 0; chunk < 4; ++chunk) {
    const int cbase = chunk << 8;
    const int n0    = cbase + (wave << 4);

    Frag bfn[4];
    if (chunk < 3) {     // prefetch next chunk's B tile; hidden under top-K
      loadB4(bfn, xnb + (size_t)(n0 + 256 + l16) * C_ + hp * 16);
    }

    v8f acc = {};
#pragma unroll
    for (int j = 0; j < 4; ++j) {
      acc = __builtin_amdgcn_wmma_f32_16x16x32_f16(false, a[j].h, false,
                                                   bf[j].h, (short)0, acc,
                                                   false, false);
    }
    const int gn = n0 + l16;
#pragma unroll
    for (int r = 0; r < 8; ++r) {
      const int mr = r + hp * 8;
      float v = acc[r];
      if (m0 + mr == gn) v = INF_;   // self-similarity -> first neighbor
      strip[mr][(n0 - cbase) + l16] = v;
    }
    __syncthreads();

    // --- top-9 over this chunk for row == wave ---
    {
      const int row = wave;
      float lv[8]; int li[8];
#pragma unroll
      for (int p = 0; p < 8; ++p) { lv[p] = NEGF; li[p] = 0x7fffffff; }
#pragma unroll
      for (int j = 0; j < 8; ++j) {
        const int   col = lane + j * 32;
        const float v   = strip[row][col];
        const int   gc  = cbase + col;
        if (v > lv[7]) {
          lv[7] = v; li[7] = gc;
#pragma unroll
          for (int p = 7; p > 0; --p) {
            if (lv[p] > lv[p - 1]) {
              const float tv = lv[p]; lv[p] = lv[p - 1]; lv[p - 1] = tv;
              const int   ti = li[p]; li[p] = li[p - 1]; li[p - 1] = ti;
            }
          }
        }
      }
#pragma unroll
      for (int t = 0; t < 9; ++t) {
        float mv = lv[0]; int mc = li[0];
#pragma unroll
        for (int off = 16; off > 0; off >>= 1) {
          const float ov = __shfl_xor(mv, off, 32);
          const int   oc = __shfl_xor(mc, off, 32);
          if (ov > mv || (ov == mv && oc < mc)) { mv = ov; mc = oc; }
        }
        if (lane == 0) { cwv[row][t] = mv; cwi[row][t] = mc; }
        if (li[0] == mc) {   // winner pops; columns are unique across lanes
#pragma unroll
          for (int p = 0; p < 7; ++p) { lv[p] = lv[p + 1]; li[p] = li[p + 1]; }
          lv[7] = NEGF; li[7] = 0x7fffffff;
        }
      }
      if (lane == 0) {   // merge two sorted 9-lists (running has lower cols)
        float nv[9]; int ni[9];
        int i = 0, j = 0;
#pragma unroll
        for (int t = 0; t < 9; ++t) {
          const float av = rvs[row][i]; const int ai = ris[row][i];
          const float bv = cwv[row][j]; const int bi = cwi[row][j];
          const bool takeA = (av > bv) || (av == bv && ai <= bi);
          if (takeA) { nv[t] = av; ni[t] = ai; ++i; }
          else       { nv[t] = bv; ni[t] = bi; ++j; }
        }
#pragma unroll
        for (int t = 0; t < 9; ++t) { rvs[row][t] = nv[t]; ris[row][t] = ni[t]; }
      }
    }
    __syncthreads();

#pragma unroll
    for (int j = 0; j < 4; ++j) bf[j] = bfn[j];   // rotate double buffer
  }

  if (tid < 256) {
    const int row = tid >> 4, kk = tid & 15;
    idxws[((size_t)(b * N_ + m0 + row)) * KPAD + kk] = ris[row][kk];
  }
}

// ---------------------------------------------------------------------------
// Kernel 3: neighbor gather + conv contraction as GEMM via WMMA.
// Block = 256 threads (8 waves); wave w owns output tile (16 tokens x 16 o's).
//
// The gathered A data (16 tokens x 9 taps = 144 rows x 256 B = 36 KB) is
// identical for all 8 waves, so it is staged ONCE per block into LDS via
// global_load_async_to_lds_b128 (ASYNCcnt), with rows padded to 272 B so the
// 16-lane ds_load_b128 A-frag reads rotate across banks (68-dword stride).
// Inner loop is software-pipelined: frags for kk+1 load while kk's WMMA runs.
// ---------------------------------------------------------------------------
#define AROW 136   // padded row length in halves (272 B)

__device__ __forceinline__ void loadA_lds(Frag& f, const _Float16* ab,
                                          int l16, int hp, int kk) {
  const int kcv = kk >> 2;          // conv tap 0..8
  const int cb  = (kk & 3) << 5;    // channel base within the tap
  const _Float16* p = ab + (l16 * K_ + kcv) * AROW + cb + hp * 8;
  f.u[0] = *(const v4u*)p;
  f.u[1] = *(const v4u*)(p + 16);
}

__device__ __forceinline__ void loadB_g(Frag& f, const _Float16* wsw,
                                        int wave, int lane, int kk) {
  const _Float16* br = wsw + ((size_t)((kk * 8 + wave) * 32 + lane)) * 16;
  f.u[0] = *(const v4u*)br;
  f.u[1] = *(const v4u*)(br + 8);
}

__global__ __launch_bounds__(256) void k_gather_conv(
    const _Float16* __restrict__ xf, const _Float16* __restrict__ wsw,
    const int* __restrict__ idxws, float* __restrict__ out) {
  __shared__ int tki[16 * K_];                       // 144 neighbor ids
  __shared__ __align__(16) _Float16 abuf[144 * AROW]; // 38.25 KB gathered rows

  const int tid  = threadIdx.x;
  const int wave = tid >> 5;
  const int lane = tid & 31;
  const int b    = blockIdx.x >> 6;
  const int n0   = (blockIdx.x & 63) << 4;

  if (tid < 16 * K_) {
    const int row = tid / K_, k = tid % K_;
    tki[tid] = idxws[((size_t)(b * N_ + n0 + row)) * KPAD + k];
  }
  __syncthreads();

  // Stage all 144 gathered token rows into LDS with async b128 copies:
  // 144 rows * 16 sixteen-byte parts = 2304 transfers / 256 threads = 9 each.
  const _Float16* xb = xf + ((size_t)b * N_) * C_;
#pragma unroll
  for (int i = 0; i < 9; ++i) {
    const int u    = i * 256 + tid;
    const int r    = u >> 4;        // gathered row 0..143
    const int part = u & 15;        // 16B chunk within the 256B row
    const int t    = tki[r];
    const _Float16* g = xb + (size_t)t * C_ + part * 8;
    const unsigned int l =
        (unsigned int)(uintptr_t)&abuf[r * AROW + part * 8];
    async_g2l_b128(l, g);
  }
  wait_asynccnt0();
  __syncthreads();

  const int hp  = lane >> 4;
  const int l16 = lane & 15;
  v8f acc = {};

  Frag af, bf;
  loadA_lds(af, abuf, l16, hp, 0);
  loadB_g(bf, wsw, wave, lane, 0);
#pragma unroll
  for (int kk = 0; kk < 36; ++kk) {
    Frag afn, bfn;
    if (kk < 35) {                       // prefetch next iteration's frags
      loadA_lds(afn, abuf, l16, hp, kk + 1);
      loadB_g(bfn, wsw, wave, lane, kk + 1);
    }
    acc = __builtin_amdgcn_wmma_f32_16x16x32_f16(false, af.h, false, bf.h,
                                                 (short)0, acc, false, false);
    af = afn; bf = bfn;
  }

  // D tile: M = token, N = output channel. Store to out (B, O, N).
  const int o = (wave << 4) + l16;
  float* ob = out + ((size_t)(b * O_ + o)) * N_ + n0;
#pragma unroll
  for (int r = 0; r < 8; ++r) {
    ob[r + hp * 8] = acc[r];
  }
}

// ---------------------------------------------------------------------------
// Launch. Workspace layout (all offsets 16B-aligned):
//   xn16  @ 0        : 8 MB   (B*N*C f16, normalized, token-major)
//   xf16  @ 8  MB    : 8 MB   (B*N*C f16, raw, token-major)
//   wsw   @ 16 MB    : 288 KB (swizzled weights, WMMA-B order)
//   idx   @ 16.5 MB  : 2 MB   (B*N*KPAD int32 neighbor indices)
// Total ~18.5 MB.
// ---------------------------------------------------------------------------
extern "C" void kernel_launch(void* const* d_in, const int* in_sizes, int n_in,
                              void* d_out, int out_size, void* d_ws, size_t ws_size,
                              hipStream_t stream) {
  (void)in_sizes; (void)n_in; (void)out_size; (void)ws_size;
  const float* x  = (const float*)d_in[0];
  const float* Wg = (const float*)d_in[1];
  float* out = (float*)d_out;
  char* ws = (char*)d_ws;

  _Float16* xn  = (_Float16*)(ws);
  _Float16* xfh = (_Float16*)(ws + (size_t)8 * 1024 * 1024);
  _Float16* wsw = (_Float16*)(ws + (size_t)16 * 1024 * 1024);
  int*      idx = (int*)     (ws + (size_t)16 * 1024 * 1024 + 512 * 1024);

  hipLaunchKernelGGL(k_normalize,   dim3(512),  dim3(256), 0, stream, x, xn, xfh);
  hipLaunchKernelGGL(k_wswizzle,    dim3(576),  dim3(256), 0, stream, Wg, wsw);
  hipLaunchKernelGGL(k_sim_topk,    dim3(2048), dim3(512), 0, stream, xn, idx);
  hipLaunchKernelGGL(k_gather_conv, dim3(2048), dim3(256), 0, stream, xfh, wsw, idx, out);
}
